// IterativeDecimator_69578470195871
// MI455X (gfx1250) — compile-verified, matrix operand order checked by reading, not run on hardware
//
#include <hip/hip_runtime.h>

typedef __attribute__((ext_vector_type(4)))  float    v4f;
typedef __attribute__((ext_vector_type(8)))  _Float16 v8h;
typedef __attribute__((ext_vector_type(16))) _Float16 v16h;
typedef __attribute__((ext_vector_type(8)))  float    v8f;

#define DFEAT 128   // node feature dim
#define HID   32    // MLP hidden
#define KCL   16    // clusters
#define TNODES 64   // nodes per block

// Build a 16-half WMMA operand from two contiguous 16-byte LDS chunks.
// Elements 0..7  <- K = base + 8*hi .. +7      (VGPRs 0..3)
// Elements 8..15 <- K = base + 16 + 8*hi .. +7 (VGPRs 4..7)
// Caller passes p = &buf[row_base + 8*hi]; second chunk is p+16.
__device__ __forceinline__ v16h ld16(const _Float16* p) {
  v8h lo = *(const v8h*)p;
  v8h hh = *(const v8h*)(p + 16);
  return __builtin_shufflevector(lo, hh, 0, 1, 2, 3, 4, 5, 6, 7,
                                 8, 9, 10, 11, 12, 13, 14, 15);
}

__device__ __forceinline__ float half_red_max(float x) {
  x = fmaxf(x, __shfl_xor(x, 1, 32));
  x = fmaxf(x, __shfl_xor(x, 2, 32));
  x = fmaxf(x, __shfl_xor(x, 4, 32));
  x = fmaxf(x, __shfl_xor(x, 8, 32));
  return x;
}
__device__ __forceinline__ float half_red_sum(float x) {
  x += __shfl_xor(x, 1, 32);
  x += __shfl_xor(x, 2, 32);
  x += __shfl_xor(x, 4, 32);
  x += __shfl_xor(x, 8, 32);
  return x;
}

__global__ __launch_bounds__(256)
void dp_fused(const float* __restrict__ X, const float* __restrict__ W1,
              const float* __restrict__ B1, const float* __restrict__ W2,
              const float* __restrict__ B2, float* __restrict__ out,
              int graphShift, long long assignOff) {
  __shared__ _Float16 xf16[TNODES * DFEAT];  // 16 KB [node][feat]
  __shared__ _Float16 xT[DFEAT * TNODES];    // 16 KB [feat][node]
  __shared__ _Float16 w1t[HID * DFEAT];      //  8 KB [col][k]
  __shared__ _Float16 w2t[KCL * HID];        //  1 KB [col][k]
  __shared__ _Float16 hf[TNODES * HID];      //  4 KB [node][hid]
  __shared__ _Float16 asnT[KCL * TNODES];    //  2 KB [cluster][node]

  const int tid = threadIdx.x;
  const long long n0 = (long long)blockIdx.x * TNODES;
  const int g = blockIdx.x >> graphShift;

  // ---- stage 0: global -> LDS, f16 conversion; X stored both orientations.
  // X is streamed exactly once -> non-temporal loads keep it out of L2.
  {
    const v4f* xsrc = (const v4f*)(X + n0 * DFEAT);
    for (int i = tid; i < (TNODES * DFEAT) / 4; i += 256) {
      v4f f = __builtin_nontemporal_load(&xsrc[i]);
      int node = i >> 5;            // 32 float4 per 128-feat row
      int feat = (i & 31) * 4;
      _Float16 h0 = (_Float16)f.x, h1 = (_Float16)f.y;
      _Float16 h2 = (_Float16)f.z, h3 = (_Float16)f.w;
      xf16[node * DFEAT + feat + 0] = h0;
      xf16[node * DFEAT + feat + 1] = h1;
      xf16[node * DFEAT + feat + 2] = h2;
      xf16[node * DFEAT + feat + 3] = h3;
      xT[(feat + 0) * TNODES + node] = h0;
      xT[(feat + 1) * TNODES + node] = h1;
      xT[(feat + 2) * TNODES + node] = h2;
      xT[(feat + 3) * TNODES + node] = h3;
    }
    for (int i = tid; i < DFEAT * HID; i += 256) {  // W1 [k][c] -> w1t [c][k]
      int k = i >> 5, c = i & 31;
      w1t[c * DFEAT + k] = (_Float16)W1[i];
    }
    for (int i = tid; i < HID * KCL; i += 256) {    // W2 [k][c] -> w2t [c][k]
      int k = i >> 4, c = i & 15;
      w2t[c * HID + k] = (_Float16)W2[i];
    }
  }
  __syncthreads();

  const int w    = tid >> 5;        // wave id 0..7
  const int lane = tid & 31;
  const int m    = lane & 15;       // row (A) / col (B, C/D) within tile
  const int hi   = lane >> 4;       // half-wave select
  const int kh   = 8 * hi;

  // ---- stage 1: h = relu(X @ W1 + b1)
  // 8 wave-tiles: node tile t = w>>1 (16 rows), col tile cb = (w&1)*16
  {
    const int t  = w >> 1;
    const int cb = (w & 1) * 16;
    const int m0 = t * 16;
    v8f c0 = {};
#pragma unroll
    for (int ks = 0; ks < 4; ++ks) {
      v16h a = ld16(&xf16[(m0 + m) * DFEAT + ks * 32 + kh]);
      v16h b = ld16(&w1t[(cb + m) * DFEAT + ks * 32 + kh]);
      c0 = __builtin_amdgcn_wmma_f32_16x16x32_f16(false, a, false, b, (short)0, c0, false, false);
    }
    float bias = B1[cb + m];
#pragma unroll
    for (int v = 0; v < 8; ++v) {
      int row = m0 + v + 8 * hi;
      float h = c0[v] + bias;
      h = h > 0.f ? h : 0.f;
      hf[row * HID + cb + m] = (_Float16)h;
    }
  }
  __syncthreads();

  // ---- stage 2+3: logits = h @ W2 + b2, softmax, assignments.
  // Only 4 node-tiles exist; waves 4..7 recompute tile (w&3) so that the
  // WMMA and shuffles always run with EXEC all-ones (ISA requirement).
  // Only the stores are predicated on (w < 4).
  {
    const int m0 = (w & 3) * 16;
    v16h a2 = ld16(&hf[(m0 + m) * HID + kh]);
    v16h b2 = ld16(&w2t[m * HID + kh]);
    v8f zero = {};
    v8f lg = __builtin_amdgcn_wmma_f32_16x16x32_f16(false, a2, false, b2, (short)0, zero, false, false);
    float bias2 = B2[m];
    float* outAssign = out + assignOff;
#pragma unroll
    for (int v = 0; v < 8; ++v) {
      float x = lg[v] + bias2;
      float r = half_red_max(x);
      float e = __expf(x - r);
      float s = half_red_sum(e);
      float av = e / s;
      int row = m0 + v + 8 * hi;
      if (w < 4) {
        // streamed once, never re-read by this kernel -> non-temporal
        __builtin_nontemporal_store(av, &outAssign[(n0 + row) * KCL + m]);
        asnT[m * TNODES + row] = (_Float16)av;   // [cluster][local node]
      }
    }
  }
  __syncthreads();

  // ---- stage 4: coarse_g += A^T X  (K = 64 local nodes, 2 WMMA steps)
  // per wave: one [16 clusters x 16 feats] tile, feats f0..f0+15
  {
    const int f0 = w * 16;
    v8f acc = {};
#pragma unroll
    for (int ks = 0; ks < 2; ++ks) {
      v16h a3 = ld16(&asnT[m * TNODES + ks * 32 + kh]);
      v16h b3 = ld16(&xT[(f0 + m) * TNODES + ks * 32 + kh]);
      acc = __builtin_amdgcn_wmma_f32_16x16x32_f16(false, a3, false, b3, (short)0, acc, false, false);
    }
#pragma unroll
    for (int v = 0; v < 8; ++v) {
      int cl = v + 8 * hi;
      unsafeAtomicAdd(&out[((long long)(g * KCL + cl)) * DFEAT + f0 + m], acc[v]);
    }
  }
}

// Zero the coarse region (atomics accumulate into it) and emit the
// deterministic sender/receiver/edge arrays.
__global__ void dp_init(float* __restrict__ out, long long coarseCnt,
                        long long sendOff, long long E) {
  long long i = (long long)blockIdx.x * blockDim.x + threadIdx.x;
  if (i < coarseCnt) { out[i] = 0.f; return; }
  long long e = i - coarseCnt;
  if (e >= 3 * E) return;
  float* p = out + sendOff;
  if (e < E) {                       // senders: g*16 + s, s repeated
    long long idx = e;
    int gg = (int)(idx >> 8);
    int s  = (int)((idx & 255) >> 4);
    p[e] = (float)(gg * KCL + s);
  } else if (e < 2 * E) {            // receivers: g*16 + r, r tiled
    long long idx = e - E;
    int gg = (int)(idx >> 8);
    int r  = (int)(idx & 15);
    p[e] = (float)(gg * KCL + r);
  } else {                           // edges: ones
    p[e] = 1.0f;
  }
}

extern "C" void kernel_launch(void* const* d_in, const int* in_sizes, int n_in,
                              void* d_out, int out_size, void* d_ws, size_t ws_size,
                              hipStream_t stream) {
  const float* X  = (const float*)d_in[0];
  const float* W1 = (const float*)d_in[1];
  const float* B1 = (const float*)d_in[2];
  const float* W2 = (const float*)d_in[3];
  const float* B2 = (const float*)d_in[4];
  // d_in[5] = n_node (uniform P per graph in this workload)

  const int N = in_sizes[0] / DFEAT;       // 262144
  const int G = in_sizes[5];               // 256
  const int P = N / G;                     // 1024
  const int blocksPerGraph = P / TNODES;   // 16 (power of two)
  const int graphShift = __builtin_ctz(blocksPerGraph);

  const long long coarseCnt = (long long)G * KCL * DFEAT;  // 524288
  const long long assignCnt = (long long)N * KCL;          // 4194304
  const long long E         = (long long)G * KCL * KCL;    // 65536
  float* out = (float*)d_out;

  const long long initTotal = coarseCnt + 3 * E;
  dp_init<<<(int)((initTotal + 255) / 256), 256, 0, stream>>>(
      out, coarseCnt, coarseCnt + assignCnt, E);

  dp_fused<<<N / TNODES, 256, 0, stream>>>(
      X, W1, B1, W2, B2, out, graphShift, coarseCnt);
}
